// NTXentLoss_11682311045307
// MI455X (gfx1250) — matrix operand order repaired and checked by reading, not compiled, not run on hardware
//
#include <hip/hip_runtime.h>
#include <hip/hip_bf16.h>
#include <math.h>

// ---------------------------------------------------------------------------
// NT-Xent loss, fused for MI455X (gfx1250, wave32, WMMA 16x16x32 f16).
//   z  = concat(z_i, z_j)            [8192 x 256] f32
//   zn = z / max(||z||, 1e-8)        stored f16 in ws
//   sim = (zn @ zn^T) * 2            never materialized
//   loss = mean_r( log(sum_{c!=r} exp(sim[r,c])) - sim[r, r^4096] )
// Logits are bounded (|sim| <= 2) so no max-tracking is needed: the softmax
// denominator is a plain sum of exp -> associative -> column-split + atomics.
//
// B tiles (= zn rows, X @ X^T symmetry) are staged cooperatively into LDS
// and double-buffered: all 8 waves of a block consume the same tile stream,
// so LDS staging cuts VMEM traffic 8x and decouples L2 latency from WMMA.
// K-accumulation uses two independent WMMA chains (even/odd K) so each wave
// keeps two WMMAs in flight instead of one serialized C-chain.
// ---------------------------------------------------------------------------

#define BD    4096          // B
#define N2    8192          // 2B
#define DDIM  256           // D
#define RPB   128           // rows per block (8 waves * 16 rows)
#define CSPLIT 4            // column splits (grid.y)
#define CCHUNK (N2 / CSPLIT)
#define TEMP_INV 2.0f       // 1 / temperature
#define TILE_H (16 * DDIM)  // halves per 16-row B tile (8 KB)

typedef __attribute__((ext_vector_type(16))) _Float16 v16h;
typedef __attribute__((ext_vector_type(8)))  _Float16 v8h;
typedef __attribute__((ext_vector_type(8)))  float    v8f;

union AFrag { v16h v; v8h h[2]; };

// ---------------------------------------------------------------- zero accums
__global__ __launch_bounds__(256) void ntxent_zero(float* __restrict__ acc,
                                                   float* __restrict__ out) {
  const int i = blockIdx.x * 256 + threadIdx.x;
  if (i < 2 * N2) acc[i] = 0.0f;
  if (i == 0) out[0] = 0.0f;
}

// ------------------------------------------------- normalize rows, f32 -> f16
// One wave per row: 32 lanes x 8 elements. Wave-wide shfl_xor sum-of-squares.
__global__ __launch_bounds__(256) void ntxent_normalize(
    const float* __restrict__ zi, const float* __restrict__ zj,
    _Float16* __restrict__ zn) {
  const int wave = threadIdx.x >> 5;
  const int lane = threadIdx.x & 31;
  const int row  = blockIdx.x * 8 + wave;
  const float* src = (row < BD) ? (zi + (size_t)row * DDIM)
                                : (zj + (size_t)(row - BD) * DDIM);
  const float4 x0 = *(const float4*)(src + lane * 8);
  const float4 x1 = *(const float4*)(src + lane * 8 + 4);
  float ss = x0.x * x0.x + x0.y * x0.y + x0.z * x0.z + x0.w * x0.w
           + x1.x * x1.x + x1.y * x1.y + x1.z * x1.z + x1.w * x1.w;
#pragma unroll
  for (int m = 1; m < 32; m <<= 1) ss += __shfl_xor(ss, m, 32);
  const float scale = 1.0f / fmaxf(sqrtf(ss), 1e-8f);
  v8h o;
  o[0] = (_Float16)(x0.x * scale); o[1] = (_Float16)(x0.y * scale);
  o[2] = (_Float16)(x0.z * scale); o[3] = (_Float16)(x0.w * scale);
  o[4] = (_Float16)(x1.x * scale); o[5] = (_Float16)(x1.y * scale);
  o[6] = (_Float16)(x1.z * scale); o[7] = (_Float16)(x1.w * scale);
  *(v8h*)(zn + (size_t)row * DDIM + lane * 8) = o;
}

// ------------------- fused zn@zn^T (WMMA f16, LDS-staged B) + softmax epilogue
// Block: 8 waves. Wave w owns rows [r0, r0+16). All waves share the same
// 16-column B tiles, staged into a double-buffered LDS tile by all 256
// threads (32 B each). Next tile's global load is issued before the WMMAs
// and ds-stored after them, hiding L2 latency behind compute.
__global__ __launch_bounds__(256) void ntxent_main(
    const _Float16* __restrict__ zn,
    float* __restrict__ s_acc,      // [N2] sum of exp(sim) (excl. diagonal)
    float* __restrict__ pos_acc) {  // [N2] positive-pair logit
  __shared__ __align__(32) _Float16 lds[2 * TILE_H];   // 2 x 8 KB

  const int tid  = threadIdx.x;
  const int wave = tid >> 5;
  const int lane = tid & 31;
  const int lo   = lane & 15;       // column-within-tile / A row-within-tile
  const int hi   = lane >> 4;       // half-wave select
  const int r0   = blockIdx.x * RPB + wave * 16;
  const int c_begin = blockIdx.y * CCHUNK;
  const int c_end   = c_begin + CCHUNK;

  // Staging map: thread copies 32 B of the 8 KB tile.
  const int srow = tid >> 4;          // 0..15 tile row
  const int scol = (tid & 15) * 16;   // halves offset within row

  // A fragments resident for the whole column sweep: 8 x v16h = 64 VGPRs.
  // f16 A 16x32 layout: lane (lo) = row; halves[0..15] = K contiguous
  // at offset hi*16 (matches B).
  v16h a[8];
  {
    const _Float16* arow = zn + (size_t)(r0 + lo) * DDIM;
#pragma unroll
    for (int k = 0; k < 8; ++k) {
      AFrag f;
      f.h[0] = *(const v8h*)(arow + k * 32 + hi * 16);
      f.h[1] = *(const v8h*)(arow + k * 32 + hi * 16 + 8);
      a[k] = f.v;
    }
  }

  float s[8], pos[8];
#pragma unroll
  for (int v = 0; v < 8; ++v) { s[v] = 0.0f; pos[v] = 0.0f; }

  // Prologue: stage first tile into buffer 0.
  *(v16h*)(lds + srow * DDIM + scol) =
      *(const v16h*)(zn + (size_t)(c_begin + srow) * DDIM + scol);
  __syncthreads();

  int buf = 0;
#pragma unroll 2
  for (int c0 = c_begin; c0 < c_end; c0 += 16) {
    const bool has_next = (c0 + 16 < c_end);   // uniform
    // Early-issue next tile's global load (latency hidden behind WMMAs).
    v16h stage;
    if (has_next)
      stage = *(const v16h*)(zn + (size_t)(c0 + 16 + srow) * DDIM + scol);

    // f16 B 32x16 operand as loaded here (k order chosen to match A): lane (lo) = column; halves[0..15] = K
    // contiguous at offset hi*16.
    const _Float16* bbase = lds + buf * TILE_H + lo * DDIM + hi * 16;
    v16h b[8];
#pragma unroll
    for (int k = 0; k < 8; ++k) b[k] = *(const v16h*)(bbase + k * 32);

    // Two independent accumulator chains (even/odd K) -> 2 WMMAs in flight.
    v8f ce = {}, co = {};
#pragma unroll
    for (int k = 0; k < 8; k += 2) {
      ce = __builtin_amdgcn_wmma_f32_16x16x32_f16(
               false, a[k],     false, b[k],     (short)0, ce, false, false);
      co = __builtin_amdgcn_wmma_f32_16x16x32_f16(
               false, a[k + 1], false, b[k + 1], (short)0, co, false, false);
    }

    // Epilogue: C layout lane(lo)=col, VGPR v -> row v + hi*8.
    const int col = c0 + lo;
#pragma unroll
    for (int v = 0; v < 8; ++v) {
      const int row = r0 + v + hi * 8;
      const float sim = (ce[v] + co[v]) * TEMP_INV;
      s[v] += (col == row) ? 0.0f : __expf(sim);       // diagonal masked
      if (col == (row ^ BD)) pos[v] = sim;             // positive pair
    }

    // Commit staged tile to the other buffer; one barrier per tile.
    if (has_next)
      *(v16h*)(lds + (buf ^ 1) * TILE_H + srow * DDIM + scol) = stage;
    __syncthreads();
    buf ^= 1;
  }

  // Row reduction across the 16 lanes of each half-wave (xor masks stay
  // inside each 16-lane group).
#pragma unroll
  for (int m = 1; m < 16; m <<= 1) {
#pragma unroll
    for (int v = 0; v < 8; ++v) {
      s[v]   += __shfl_xor(s[v], m, 32);
      pos[v] += __shfl_xor(pos[v], m, 32);
    }
  }
  if (lo == 0) {
#pragma unroll
    for (int v = 0; v < 8; ++v) {
      const int row = r0 + v + hi * 8;
      atomicAdd(&s_acc[row], s[v]);     // combine column splits
      atomicAdd(&pos_acc[row], pos[v]); // only one split contributes != 0
    }
  }
}

// ----------------------------------------------------- finalize: scalar loss
__global__ __launch_bounds__(256) void ntxent_finalize(
    const float* __restrict__ s_acc, const float* __restrict__ pos_acc,
    float* __restrict__ out) {
  __shared__ float red[256];
  const int r = blockIdx.x * 256 + threadIdx.x;
  red[threadIdx.x] = __logf(s_acc[r]) - pos_acc[r];
  __syncthreads();
#pragma unroll
  for (int st = 128; st > 0; st >>= 1) {
    if (threadIdx.x < st) red[threadIdx.x] += red[threadIdx.x + st];
    __syncthreads();
  }
  if (threadIdx.x == 0) atomicAdd(out, red[0] * (1.0f / (float)N2));
}

// ---------------------------------------------------------------------------
extern "C" void kernel_launch(void* const* d_in, const int* in_sizes, int n_in,
                              void* d_out, int out_size, void* d_ws, size_t ws_size,
                              hipStream_t stream) {
  const float* zi = (const float*)d_in[0];
  const float* zj = (const float*)d_in[1];
  float* out = (float*)d_out;

  // ws layout: [0, 4MB) zn f16; then s_acc[N2]; then pos_acc[N2].
  _Float16* zn   = (_Float16*)d_ws;
  float* s_acc   = (float*)((char*)d_ws + (size_t)N2 * DDIM * sizeof(_Float16));
  float* pos_acc = s_acc + N2;

  ntxent_zero<<<(2 * N2 + 255) / 256, 256, 0, stream>>>(s_acc, out);
  ntxent_normalize<<<N2 / 8, 256, 0, stream>>>(zi, zj, zn);
  dim3 grid(N2 / RPB, CSPLIT);
  ntxent_main<<<grid, 256, 0, stream>>>(zn, s_acc, pos_acc);
  ntxent_finalize<<<N2 / 256, 256, 0, stream>>>(s_acc, pos_acc, out);
}
